// MultiHeadLatentAttention_60919816126968
// MI455X (gfx1250) — matrix operand order repaired and checked by reading, not compiled, not run on hardware
//
#include <hip/hip_runtime.h>

// ---------------------------------------------------------------------------
// MLA forward for MI455X (gfx1250): bf16 WMMA + async-to-LDS double buffering.
// ---------------------------------------------------------------------------

typedef __bf16 v16bf __attribute__((ext_vector_type(16)));
typedef __bf16 v8bf  __attribute__((ext_vector_type(8)));
typedef float  v8f   __attribute__((ext_vector_type(8)));

#define BB    2
#define TT    2048
#define HH    16
#define DD    128
#define DIMM  2048
#define DCC   512
#define DCQ   1536
#define DR    64
#define MM    (BB*TT)          // 4096 rows of x

// ---- CDNA5 sync helpers ----------------------------------------------------
static __device__ __forceinline__ void wait_async0() {
  asm volatile("s_wait_asynccnt 0x0" ::: "memory");
}
// Intra-wave LDS fence: wave32 is lane-synchronous and per-wave LDS ops are
// in-order, so draining DScnt (plus a scheduling barrier) makes same-wave
// cross-lane LDS communication visible without s_barrier.
static __device__ __forceinline__ void wave_lds_sync() {
  __builtin_amdgcn_wave_barrier();
  asm volatile("s_wait_dscnt 0x0" ::: "memory");
  __builtin_amdgcn_wave_barrier();
}
// Async DMA: global -> LDS, 16 bytes per lane (tracked by ASYNCcnt).
static __device__ __forceinline__ void async_g2l_b128(void* lds_ptr, const void* gptr) {
  unsigned int       ldsa = (unsigned int)(uintptr_t)lds_ptr;   // low 32 bits = LDS offset
  unsigned long long ga   = (unsigned long long)(uintptr_t)gptr;
  asm volatile("global_load_async_to_lds_b128 %0, %1, off"
               :: "v"(ldsa), "v"(ga) : "memory");
}

// ---- WMMA fragment loaders (contiguous k order, same for A and B) ------
// A (16x32 bf16, row-major, leading dim ldk): lane holds row = lane%16,
// elements 0..15 -> k = o+i, o = (lane<16)?0:16.
static __device__ __forceinline__
v16bf load_a_frag(const __bf16* A, int ldk, int row, int k0, int lane) {
  int o = (lane & 16) ? 16 : 0;
  const __bf16* p = A + (size_t)row * ldk + k0 + o;
  v8bf lo = *(const v8bf*)(p);
  v8bf hi = *(const v8bf*)(p + 8);
  return __builtin_shufflevector(lo, hi, 0,1,2,3,4,5,6,7,8,9,10,11,12,13,14,15);
}

static __device__ __forceinline__
v8f wmma_bf16(v16bf a, v16bf b, v8f c) {
  return __builtin_amdgcn_wmma_f32_16x16x32_bf16(false, a, false, b,
                                                 (short)0, c, false, false);
}

// ---- tiled WMMA GEMM: C[M,N] = A[M,K] (bf16) @ Bt[N,K] (bf16) --------------
// 256 threads = 8 waves; block tile 128(M) x 64(N); all waves share one B
// tile (64 n-rows x 64 k per stage) staged in LDS by the async-to-LDS engine
// with double buffering. 8 WMMAs per wave per barrier pair.
// OMODE: 0 = f32 row-major, 1 = bf16 row-major,
//        2 = bf16 scatter into K/Q[b,h,t,d], 3 = bf16 scatter into Vt[b,h,d,t]
template <int OMODE>
__global__ __launch_bounds__(256)
void gemm_wmma(const __bf16* __restrict__ A, const __bf16* __restrict__ Bt,
               void* __restrict__ out, int M, int N, int K) {
  __shared__ __bf16 Bs[2][64 * 64];          // 2 x 8 KB

  const int tid   = threadIdx.x;
  const int lane  = tid & 31;
  const int wave  = tid >> 5;
  const int mTile = blockIdx.y * 128 + wave * 16;
  const int nBase = blockIdx.x * 64;
  const int row   = mTile + (lane & 15);

  // cooperative async copy: thread t moves 32B of the 64x64 B stage
  const int crow = tid >> 2;                 // 0..63  (local n)
  const int cseg = (tid & 3) * 16;           // 0,16,32,48 (k elements)
  auto stageB = [&](int buf, int k0) {
    const __bf16* src = Bt + (size_t)(nBase + crow) * K + k0 + cseg;
    __bf16*       dst = &Bs[buf][crow * 64 + cseg];
    async_g2l_b128(dst, src);
    async_g2l_b128(dst + 8, src + 8);
  };

  v8f acc[4] = {v8f{0}, v8f{0}, v8f{0}, v8f{0}};

  stageB(0, 0);                              // prologue
  wait_async0();
  __syncthreads();

  const int koff = (lane & 16) ? 16 : 0;
  for (int k0 = 0; k0 < K; k0 += 64) {
    const int buf = (k0 >> 6) & 1;
    if (k0 + 64 < K) stageB(buf ^ 1, k0 + 64);   // overlaps WMMAs below

#pragma unroll
    for (int kk = 0; kk < 64; kk += 32) {
      v16bf a = load_a_frag(A, K, row, k0 + kk, lane);
#pragma unroll
      for (int c = 0; c < 4; ++c) {
        const __bf16* bp = &Bs[buf][(c * 16 + (lane & 15)) * 64 + kk + koff];
        v16bf b = *(const v16bf*)bp;
        acc[c] = wmma_bf16(a, b, acc[c]);
      }
    }

    wait_async0();                           // my slice of next buffer landed
    __syncthreads();                         // everyone's slice landed / reads done
  }

  const int mhalf = (lane & 16) ? 8 : 0;
#pragma unroll
  for (int c = 0; c < 4; ++c) {
    int n = nBase + c * 16 + (lane & 15);
#pragma unroll
    for (int j = 0; j < 8; ++j) {
      int   m   = mTile + mhalf + j;
      float val = acc[c][j];
      if (OMODE == 0) {
        ((float*)out)[(size_t)m * N + n] = val;
      } else if (OMODE == 1) {
        ((__bf16*)out)[(size_t)m * N + n] = (__bf16)val;
      } else if (OMODE == 2) {          // row=(b,t), col=(h,d) -> [b,h,t,d]
        int b = m >> 11, t = m & (TT - 1);
        int h = n >> 7,  d = n & (DD - 1);
        ((__bf16*)out)[((((size_t)b * HH + h) * TT) + t) * DD + d] = (__bf16)val;
      } else {                          // -> Vt[b,h,d,t]
        int b = m >> 11, t = m & (TT - 1);
        int h = n >> 7,  d = n & (DD - 1);
        ((__bf16*)out)[((((size_t)b * HH + h) * DD) + d) * TT + t] = (__bf16)val;
      }
    }
  }
}

// ---- helpers ---------------------------------------------------------------
__global__ void convert_f32_bf16(const float* __restrict__ in,
                                 __bf16* __restrict__ out, size_t n) {
  size_t i = (size_t)blockIdx.x * blockDim.x + threadIdx.x;
  if (i < n) out[i] = (__bf16)in[i];
}

// out[c*R + r] = in[r*C + c]  (f32[R,C] -> bf16[C,R])
__global__ void transpose_f32_bf16(const float* __restrict__ in,
                                   __bf16* __restrict__ out, int R, int C) {
  size_t i = (size_t)blockIdx.x * blockDim.x + threadIdx.x;
  if (i >= (size_t)R * C) return;
  int r = (int)(i / C), c = (int)(i % C);
  out[(size_t)c * R + r] = (__bf16)in[i];
}

// apply RoPE to rbuf (f32 [B*T,64]) and broadcast into dst[b,h,t,64..127] (all h)
__global__ void rope_fill(const float* __restrict__ rbuf,
                          const float* __restrict__ cosb,
                          const float* __restrict__ sinb,
                          __bf16* __restrict__ dst) {
  int idx = blockIdx.x * blockDim.x + threadIdx.x;      // over B*T*32
  if (idx >= BB * TT * 32) return;
  int j = idx & 31, t = (idx >> 5) & (TT - 1), b = idx >> 16;
  const float* r = rbuf + ((size_t)(b * TT + t)) * DR;
  float c = cosb[t * 32 + j], s = sinb[t * 32 + j];
  float t1 = r[j], t2 = r[j + 32];
  __bf16 o1 = (__bf16)(t1 * c - t2 * s);
  __bf16 o2 = (__bf16)(t2 * c + t1 * s);
#pragma unroll
  for (int h = 0; h < HH; ++h) {
    __bf16* p = dst + ((((size_t)b * HH + h) * TT) + t) * DD;
    p[64 + j] = o1;
    p[96 + j] = o2;
  }
}

// ---- flash attention: 4 waves/block, each wave owns one (b,h,16-row q tile).
// All synchronization is intra-wave (wave_lds_sync), so waves with different
// causal trip counts coexist in one workgroup without barrier divergence.
// launch_bounds(128,1): one block per SIMD-group -> full VGPR budget so the
// 8 K^T fragments keep distinct registers and the QK^T chain pipelines.
__global__ __launch_bounds__(128, 1)
void mla_attention(const __bf16* __restrict__ Q,   // [B,H,T,D]
                   const __bf16* __restrict__ Kb,  // [B,H,T,D]
                   const __bf16* __restrict__ Vt,  // [B,H,D,T]
                   __bf16* __restrict__ O) {       // [B,T,H*D]
  __shared__ float  S_lds[4][16][32];
  __shared__ __bf16 P_lds[4][16][32];
  __shared__ float  mrow[4][16], lrow[4][16], arow[4][16];

  const int wv   = threadIdx.x >> 5;
  const int lane = threadIdx.x & 31;
  const int g    = blockIdx.x * 4 + wv;             // global q-tile id
  const int qt   = g & (TT / 16 - 1);
  const int h    = (g >> 7) & (HH - 1);
  const int b    = g >> 11;
  const int q0   = qt * 16;

  const __bf16* Qh = Q  + (((size_t)b * HH + h) * TT) * DD;
  const __bf16* Kh = Kb + (((size_t)b * HH + h) * TT) * DD;
  const __bf16* Vh = Vt + (((size_t)b * HH + h) * DD) * TT;

  if (lane < 16) { mrow[wv][lane] = -1e30f; lrow[wv][lane] = 0.f; }
  wave_lds_sync();

  // Q fragments for all of d=0..127 (4 K-steps of 32)
  v16bf qa[4];
  const int qrow = q0 + (lane & 15);
#pragma unroll
  for (int kk = 0; kk < 4; ++kk) qa[kk] = load_a_frag(Qh, DD, qrow, kk * 32, lane);

  v8f oacc[8] = {v8f{0}, v8f{0}, v8f{0}, v8f{0}, v8f{0}, v8f{0}, v8f{0}, v8f{0}};
  const float scale = 0.08838834764831845f;   // 1/sqrt(128)
  const int   mhalf = (lane & 16) ? 8 : 0;
  const int   koff  = (lane & 16) ? 16 : 0;
  const int   sEnd  = q0 + 16;                // causal: need s <= q0+15

  for (int s0 = 0; s0 < sEnd; s0 += 32) {
    // ---- preload all 8 K^T fragments (distinct regs -> pipelined waits) ----
    v16bf kbf[2][4];
#pragma unroll
    for (int ns = 0; ns < 2; ++ns)
#pragma unroll
      for (int kk = 0; kk < 4; ++kk)
        kbf[ns][kk] = *(const v16bf*)(Kh + (size_t)(s0 + ns * 16 + (lane & 15)) * DD
                                         + kk * 32 + koff);

    // ---- scores S = Q K^T for 32 keys (two 16-col subtiles) ----
#pragma unroll
    for (int ns = 0; ns < 2; ++ns) {
      v8f sc = v8f{0};
#pragma unroll
      for (int kk = 0; kk < 4; ++kk) sc = wmma_bf16(qa[kk], kbf[ns][kk], sc);
      const int scol = s0 + ns * 16 + (lane & 15);
#pragma unroll
      for (int j = 0; j < 8; ++j) {
        int   m = mhalf + j;
        float v = sc[j] * scale;
        if (scol > q0 + m) v = -1e30f;        // causal mask
        S_lds[wv][m][ns * 16 + (lane & 15)] = v;
      }
    }
    wave_lds_sync();

    // ---- online softmax row update (one row per lane 0..15) ----
    if (lane < 16) {
      const int m = lane;
      float mx = mrow[wv][m];
#pragma unroll
      for (int c = 0; c < 32; ++c) mx = fmaxf(mx, S_lds[wv][m][c]);
      float alpha = __expf(mrow[wv][m] - mx);
      float sum = 0.f;
#pragma unroll
      for (int c = 0; c < 32; ++c) {
        float p = __expf(S_lds[wv][m][c] - mx);
        P_lds[wv][m][c] = (__bf16)p;
        sum += p;
      }
      lrow[wv][m] = lrow[wv][m] * alpha + sum;
      mrow[wv][m] = mx;
      arow[wv][m] = alpha;
    }
    wave_lds_sync();

    // ---- rescale O and accumulate P @ V (16x32 @ 32x128) ----
    float aj[8];
#pragma unroll
    for (int j = 0; j < 8; ++j) aj[j] = arow[wv][mhalf + j];
    v16bf pa = load_a_frag(&P_lds[wv][0][0], 32, lane & 15, 0, lane);
#pragma unroll
    for (int c = 0; c < 8; ++c) {
#pragma unroll
      for (int j = 0; j < 8; ++j) oacc[c][j] *= aj[j];
      const __bf16* p = Vh + (size_t)(c * 16 + (lane & 15)) * TT + s0 + koff;
      v16bf vb = *(const v16bf*)p;
      oacc[c] = wmma_bf16(pa, vb, oacc[c]);
    }
    wave_lds_sync();   // S_lds/P_lds/arow reused next iteration
  }

  // ---- normalize and store O[b,t,h*128+d] ----
#pragma unroll
  for (int c = 0; c < 8; ++c) {
    int d = c * 16 + (lane & 15);
#pragma unroll
    for (int j = 0; j < 8; ++j) {
      int m = mhalf + j;
      int t = q0 + m;
      float val = oacc[c][j] / lrow[wv][m];
      O[((size_t)b * TT + t) * (HH * DD) + h * DD + d] = (__bf16)val;
    }
  }
}

// ---------------------------------------------------------------------------
extern "C" void kernel_launch(void* const* d_in, const int* in_sizes, int n_in,
                              void* d_out, int out_size, void* d_ws, size_t ws_size,
                              hipStream_t stream) {
  const float* x       = (const float*)d_in[0];
  const float* cosb    = (const float*)d_in[1];
  const float* sinb    = (const float*)d_in[2];
  const float* w_kv    = (const float*)d_in[3];
  const float* w_kdec  = (const float*)d_in[4];
  const float* w_vdec  = (const float*)d_in[5];
  const float* w_qc    = (const float*)d_in[6];
  const float* w_qdec  = (const float*)d_in[7];
  const float* w_krope = (const float*)d_in[8];
  const float* w_qrope = (const float*)d_in[9];
  const float* w_o     = (const float*)d_in[10];
  float* out = (float*)d_out;

  char* ws = (char*)d_ws;
  size_t off = 0;
  auto take = [&](size_t elems, size_t esz) -> void* {
    void* p = ws + off;
    off = (off + elems * esz + 255) & ~(size_t)255;
    return p;
  };

  __bf16* xb      = (__bf16*)take((size_t)MM * DIMM, 2);
  __bf16* wkvT    = (__bf16*)take((size_t)DCC * DIMM, 2);
  __bf16* wkdecT  = (__bf16*)take((size_t)(HH * DD) * DCC, 2);
  __bf16* wvdecT  = (__bf16*)take((size_t)(HH * DD) * DCC, 2);
  __bf16* wqcT    = (__bf16*)take((size_t)DCQ * DIMM, 2);
  __bf16* wqdecT  = (__bf16*)take((size_t)(HH * DD) * DCQ, 2);
  __bf16* wkropeT = (__bf16*)take((size_t)DR * DIMM, 2);
  __bf16* wqropeT = (__bf16*)take((size_t)DR * DIMM, 2);
  __bf16* woT     = (__bf16*)take((size_t)DIMM * (HH * DD), 2);
  __bf16* ckv     = (__bf16*)take((size_t)MM * DCC, 2);
  __bf16* cq      = (__bf16*)take((size_t)MM * DCQ, 2);
  __bf16* Kbuf    = (__bf16*)take((size_t)BB * HH * TT * DD, 2);
  __bf16* Qbuf    = (__bf16*)take((size_t)BB * HH * TT * DD, 2);
  __bf16* Vtb     = (__bf16*)take((size_t)BB * HH * DD * TT, 2);
  __bf16* Obuf    = (__bf16*)take((size_t)MM * (HH * DD), 2);
  float*  kropeF  = (float*)take((size_t)MM * DR, 4);
  float*  qropeF  = (float*)take((size_t)MM * DR, 4);
  (void)ws_size; (void)n_in; (void)in_sizes; (void)out_size;

  auto grid1 = [](size_t n) { return dim3((unsigned)((n + 255) / 256)); };

  // 1) precision conversion + weight transposes
  convert_f32_bf16<<<grid1((size_t)MM * DIMM), 256, 0, stream>>>(x, xb, (size_t)MM * DIMM);
  transpose_f32_bf16<<<grid1((size_t)DIMM * DCC), 256, 0, stream>>>(w_kv, wkvT, DIMM, DCC);
  transpose_f32_bf16<<<grid1((size_t)DCC * (HH * DD)), 256, 0, stream>>>(w_kdec, wkdecT, DCC, HH * DD);
  transpose_f32_bf16<<<grid1((size_t)DCC * (HH * DD)), 256, 0, stream>>>(w_vdec, wvdecT, DCC, HH * DD);
  transpose_f32_bf16<<<grid1((size_t)DIMM * DCQ), 256, 0, stream>>>(w_qc, wqcT, DIMM, DCQ);
  transpose_f32_bf16<<<grid1((size_t)DCQ * (HH * DD)), 256, 0, stream>>>(w_qdec, wqdecT, DCQ, HH * DD);
  transpose_f32_bf16<<<grid1((size_t)DIMM * DR), 256, 0, stream>>>(w_krope, wkropeT, DIMM, DR);
  transpose_f32_bf16<<<grid1((size_t)DIMM * DR), 256, 0, stream>>>(w_qrope, wqropeT, DIMM, DR);
  transpose_f32_bf16<<<grid1((size_t)(HH * DD) * DIMM), 256, 0, stream>>>(w_o, woT, HH * DD, DIMM);

  // 2) latent projections
  gemm_wmma<1><<<dim3(DCC / 64, MM / 128), 256, 0, stream>>>(xb, wkvT, ckv, MM, DCC, DIMM);
  gemm_wmma<1><<<dim3(DCQ / 64, MM / 128), 256, 0, stream>>>(xb, wqcT, cq, MM, DCQ, DIMM);

  // 3) decode K, V, Q with layout-scatter stores
  gemm_wmma<2><<<dim3((HH * DD) / 64, MM / 128), 256, 0, stream>>>(ckv, wkdecT, Kbuf, MM, HH * DD, DCC);
  gemm_wmma<3><<<dim3((HH * DD) / 64, MM / 128), 256, 0, stream>>>(ckv, wvdecT, Vtb, MM, HH * DD, DCC);
  gemm_wmma<2><<<dim3((HH * DD) / 64, MM / 128), 256, 0, stream>>>(cq, wqdecT, Qbuf, MM, HH * DD, DCQ);

  // 4) RoPE branches (small GEMMs, f32 out) + broadcast into K/Q d>=64
  gemm_wmma<0><<<dim3(DR / 64, MM / 128), 256, 0, stream>>>(xb, wkropeT, kropeF, MM, DR, DIMM);
  gemm_wmma<0><<<dim3(DR / 64, MM / 128), 256, 0, stream>>>(xb, wqropeT, qropeF, MM, DR, DIMM);
  rope_fill<<<dim3(BB * TT * 32 / 256), 256, 0, stream>>>(kropeF, cosb, sinb, Kbuf);
  rope_fill<<<dim3(BB * TT * 32 / 256), 256, 0, stream>>>(qropeF, cosb, sinb, Qbuf);

  // 5) causal flash attention (4 q-tiles per block, wave-local sync only)
  mla_attention<<<dim3(BB * HH * (TT / 16) / 4), 128, 0, stream>>>(Qbuf, Kbuf, Vtb, Obuf);

  // 6) output projection (f32 into d_out)
  gemm_wmma<0><<<dim3(DIMM / 64, MM / 128), 256, 0, stream>>>(Obuf, woT, out, MM, DIMM, HH * DD);
}
